// MultiHeadAttention_38268158608116
// MI455X (gfx1250) — matrix operand order repaired and checked by reading, not compiled, not run on hardware
//
#include <hip/hip_runtime.h>

#define EMBED 1024
#define NHEAD 16
#define HDIM  64
#define BSZ   4
#define SEQ   2048
#define MTOT  (BSZ * SEQ)                        // 8192 rows total
#define HSZ   ((size_t)BSZ * NHEAD * SEQ * HDIM) // elems per Q/K/Vt buffer

typedef __attribute__((ext_vector_type(16))) __bf16 v16bf;
typedef __attribute__((ext_vector_type(8)))  __bf16 v8bf;
typedef __attribute__((ext_vector_type(8)))  float  v8f;
typedef __attribute__((ext_vector_type(4)))  float  v4f;

__device__ __forceinline__ v8f wmma_bf16(v16bf a, v16bf b, v8f c) {
  // D = A(16x32 bf16) x B(32x16 bf16) + C(16x16 f32)
  return __builtin_amdgcn_wmma_f32_16x16x32_bf16(false, a, false, b,
                                                 (short)0, c, false, false);
}

__device__ __forceinline__ v16bf cvt_frag_f32(v4f a0, v4f a1, v4f a2, v4f a3) {
  v16bf f;
  #pragma unroll
  for (int i = 0; i < 4; ++i) {
    f[i]      = (__bf16)a0[i];
    f[4 + i]  = (__bf16)a1[i];
    f[8 + i]  = (__bf16)a2[i];
    f[12 + i] = (__bf16)a3[i];
  }
  return f;
}

// ---------------------------------------------------------------------------
// Projection GEMM: out = X(fp32)[M,K] @ W(fp32)[N,K]^T + bias, stored bf16.
// MODE 0: out[b][h][s][d]   (Q, K layout)
// MODE 1: out[b][h][d][s]   (V transposed per head, for PV WMMA B-operand)
// One wave computes a 32(M) x 64(N) tile: A-frags reused over 4 N-subtiles,
// B-frags reused over 2 M-subtiles -> 8 WMMA per 24 16B-loads per k-step.
// 512 blocks x 8 waves = 4096 tiles.
// ---------------------------------------------------------------------------
template <int MODE>
__global__ __launch_bounds__(256) void proj_gemm(const float* __restrict__ X,
                                                 const float* __restrict__ W,
                                                 const float* __restrict__ bias,
                                                 __bf16* __restrict__ out,
                                                 float outScale) {
  const int lane = threadIdx.x & 31;
  const int wid  = threadIdx.x >> 5;
  const int lrow = lane & 15;
  const int lhi  = lane >> 4;
  const int gid  = blockIdx.x * 8 + wid;
  const int m0   = (gid >> 4) * 32;   // 256 m-tiles of 32 rows
  const int n0   = (gid & 15) * 64;   // 16 n-tiles of width 64

  const float* arow0 = X + (size_t)(m0 + lrow) * EMBED + lhi * 16;
  const float* arow1 = arow0 + (size_t)16 * EMBED;

  v8f acc[2][4];
  #pragma unroll
  for (int t = 0; t < 4; ++t) {
    const float bv = bias[n0 + t * 16 + lrow];
    #pragma unroll
    for (int r = 0; r < 8; ++r) { acc[0][t][r] = bv; acc[1][t][r] = bv; }
  }

  for (int k0 = 0; k0 < EMBED; k0 += 32) {
    v16bf af0 = cvt_frag_f32(*(const v4f*)(arow0 + k0),
                             *(const v4f*)(arow0 + k0 + 4),
                             *(const v4f*)(arow0 + k0 + 8),
                             *(const v4f*)(arow0 + k0 + 12));
    v16bf af1 = cvt_frag_f32(*(const v4f*)(arow1 + k0),
                             *(const v4f*)(arow1 + k0 + 4),
                             *(const v4f*)(arow1 + k0 + 8),
                             *(const v4f*)(arow1 + k0 + 12));
    #pragma unroll
    for (int t = 0; t < 4; ++t) {
      const float* wrow = W + (size_t)(n0 + t * 16 + lrow) * EMBED + k0 + lhi * 16;
      v16bf bf = cvt_frag_f32(*(const v4f*)(wrow),
                              *(const v4f*)(wrow + 4),
                              *(const v4f*)(wrow + 8),
                              *(const v4f*)(wrow + 12));
      acc[0][t] = wmma_bf16(af0, bf, acc[0][t]);
      acc[1][t] = wmma_bf16(af1, bf, acc[1][t]);
    }
  }

  #pragma unroll
  for (int mi = 0; mi < 2; ++mi) {
    #pragma unroll
    for (int t = 0; t < 4; ++t) {
      const int n = n0 + t * 16 + lrow;
      const int h = n >> 6, d = n & (HDIM - 1);
      #pragma unroll
      for (int r = 0; r < 8; ++r) {
        const int m = m0 + mi * 16 + r + 8 * lhi;
        const int b = m >> 11, s = m & (SEQ - 1);
        const __bf16 val = (__bf16)(acc[mi][t][r] * outScale);
        if (MODE == 0)
          out[((size_t)(b * NHEAD + h) * SEQ + s) * HDIM + d] = val;
        else
          out[((size_t)(b * NHEAD + h) * HDIM + d) * SEQ + s] = val;
      }
    }
  }
}

// ---------------------------------------------------------------------------
// Flash attention: one wave owns a 32-query tile of one (b,h); streams 32
// keys per iteration with online softmax. K/V fragments are reused across the
// two 16-query sub-tiles -> 16 WMMA per iteration. Q pre-scaled by Dh^-0.5.
// Q,K: [B,H,S,Dh] bf16; Vt: [B,H,Dh,S] bf16; Aout: [B,S,E] bf16.
// 512 blocks x 8 waves = 4096 q-tiles.
// ---------------------------------------------------------------------------
__global__ __launch_bounds__(256) void flash_attn(const __bf16* __restrict__ Q,
                                                  const __bf16* __restrict__ K,
                                                  const __bf16* __restrict__ Vt,
                                                  __bf16* __restrict__ Aout) {
  __shared__ __bf16 Pl[8][32][40];   // per-wave P tile, padded rows (80B stride)

  const int lane = threadIdx.x & 31;
  const int wid  = threadIdx.x >> 5;
  const int lrow = lane & 15;
  const int lhi  = lane >> 4;
  const int gid  = blockIdx.x * 8 + wid;
  const int bh   = gid >> 6;          // 0..63
  const int q0   = (gid & 63) * 32;   // query tile base (32 rows)

  const __bf16* Qh = Q  + (size_t)bh * SEQ * HDIM;
  const __bf16* Kh = K  + (size_t)bh * SEQ * HDIM;
  const __bf16* Vh = Vt + (size_t)bh * HDIM * SEQ;

  // Q A-fragments: [mi][dh-half], kept live for whole key loop.
  v16bf aq[2][2];
  #pragma unroll
  for (int mi = 0; mi < 2; ++mi)
    #pragma unroll
    for (int s = 0; s < 2; ++s) {
      const __bf16* qp = Qh + (size_t)(q0 + mi * 16 + lrow) * HDIM + s * 32 + lhi * 16;
      v8bf qa = *(const v8bf*)qp;
      v8bf qb = *(const v8bf*)(qp + 8);
      #pragma unroll
      for (int i = 0; i < 8; ++i) { aq[mi][s][i] = qa[i]; aq[mi][s][8 + i] = qb[i]; }
    }

  v8f o[2][4];
  #pragma unroll
  for (int mi = 0; mi < 2; ++mi)
    #pragma unroll
    for (int c = 0; c < 4; ++c)
      #pragma unroll
      for (int r = 0; r < 8; ++r) o[mi][c][r] = 0.f;
  float mrow[2][8], lsum[2][8];
  #pragma unroll
  for (int mi = 0; mi < 2; ++mi)
    #pragma unroll
    for (int r = 0; r < 8; ++r) { mrow[mi][r] = -__builtin_inff(); lsum[mi][r] = 0.f; }

  for (int j0 = 0; j0 <= q0 + 31; j0 += 32) {
    // ---- scores: per key sub-tile kt load 2 K-frags, feed both q sub-tiles
    v8f sc[2][2];                     // [mi][kt]
    #pragma unroll
    for (int kt = 0; kt < 2; ++kt) {
      const __bf16* kp = Kh + (size_t)(j0 + kt * 16 + lrow) * HDIM + lhi * 16;
      v16bf kb0 = *(const v16bf*)kp;
      v16bf kb1 = *(const v16bf*)(kp + 32);
      #pragma unroll
      for (int mi = 0; mi < 2; ++mi) {
        v8f z = {};
        z = wmma_bf16(aq[mi][0], kb0, z);
        sc[mi][kt] = wmma_bf16(aq[mi][1], kb1, z);
      }
    }

    // ---- causal mask (only blocks touching the diagonal)
    if (j0 + 31 > q0) {
      #pragma unroll
      for (int mi = 0; mi < 2; ++mi)
        #pragma unroll
        for (int r = 0; r < 8; ++r) {
          const int qr = q0 + mi * 16 + r + 8 * lhi;
          if (j0 + lrow > qr)      sc[mi][0][r] = -__builtin_inff();
          if (j0 + 16 + lrow > qr) sc[mi][1][r] = -__builtin_inff();
        }
    }

    // ---- online softmax: row stats via 16-lane xor reductions
    float alpha[2][8];
    #pragma unroll
    for (int mi = 0; mi < 2; ++mi)
      #pragma unroll
      for (int r = 0; r < 8; ++r) {
        float mx = fmaxf(sc[mi][0][r], sc[mi][1][r]);
        mx = fmaxf(mx, __shfl_xor(mx, 1, 32));
        mx = fmaxf(mx, __shfl_xor(mx, 2, 32));
        mx = fmaxf(mx, __shfl_xor(mx, 4, 32));
        mx = fmaxf(mx, __shfl_xor(mx, 8, 32));
        const float mnew = fmaxf(mrow[mi][r], mx);
        alpha[mi][r] = __expf(mrow[mi][r] - mnew);
        mrow[mi][r]  = mnew;
        const float p0 = __expf(sc[mi][0][r] - mnew);
        const float p1 = __expf(sc[mi][1][r] - mnew);
        float ps = p0 + p1;
        ps += __shfl_xor(ps, 1, 32);
        ps += __shfl_xor(ps, 2, 32);
        ps += __shfl_xor(ps, 4, 32);
        ps += __shfl_xor(ps, 8, 32);
        lsum[mi][r] = lsum[mi][r] * alpha[mi][r] + ps;
        const int qr = mi * 16 + r + 8 * lhi;
        Pl[wid][qr][lrow]      = (__bf16)p0;   // C-layout -> LDS row-major
        Pl[wid][qr][16 + lrow] = (__bf16)p1;
      }
    #pragma unroll
    for (int mi = 0; mi < 2; ++mi)
      #pragma unroll
      for (int c = 0; c < 4; ++c)
        #pragma unroll
        for (int r = 0; r < 8; ++r) o[mi][c][r] *= alpha[mi][r];

    // ---- P A-fragments (16x32 each) back from LDS (same-wave DS in-order)
    v16bf pf[2];
    #pragma unroll
    for (int mi = 0; mi < 2; ++mi) {
      const __bf16* pp = &Pl[wid][mi * 16 + lrow][0];
      v8bf pa = *(const v8bf*)(pp + lhi * 16);
      v8bf pb = *(const v8bf*)(pp + lhi * 16 + 8);
      #pragma unroll
      for (int i = 0; i < 8; ++i) { pf[mi][i] = pa[i]; pf[mi][8 + i] = pb[i]; }
    }

    // ---- O += P @ V  (4 dh-chunks; each V frag reused for both q sub-tiles)
    #pragma unroll
    for (int c = 0; c < 4; ++c) {
      const __bf16* vp = Vh + (size_t)(c * 16 + lrow) * SEQ + j0 + lhi * 16;
      v16bf vf = *(const v16bf*)vp;
      o[0][c] = wmma_bf16(pf[0], vf, o[0][c]);
      o[1][c] = wmma_bf16(pf[1], vf, o[1][c]);
    }
  }

  // ---- epilogue: normalize, store [B, S, H*Dh] bf16 for the output GEMM
  const int b = bh >> 4, h = bh & 15;
  #pragma unroll
  for (int mi = 0; mi < 2; ++mi) {
    float inv[8];
    #pragma unroll
    for (int r = 0; r < 8; ++r) inv[r] = 1.0f / lsum[mi][r];
    #pragma unroll
    for (int c = 0; c < 4; ++c) {
      const int d = h * HDIM + c * 16 + lrow;
      #pragma unroll
      for (int r = 0; r < 8; ++r) {
        const int s = q0 + mi * 16 + r + 8 * lhi;
        Aout[((size_t)b * SEQ + s) * EMBED + d] = (__bf16)(o[mi][c][r] * inv[r]);
      }
    }
  }
}

// ---------------------------------------------------------------------------
// Output GEMM: out(fp32) = A(bf16)[M,K] @ Wo(fp32)[N,K]^T + bo
// Same 32x64 per-wave tiling as proj_gemm. 512 blocks x 8 waves.
// ---------------------------------------------------------------------------
__global__ __launch_bounds__(256) void out_gemm(const __bf16* __restrict__ A,
                                                const float* __restrict__ W,
                                                const float* __restrict__ bias,
                                                float* __restrict__ out) {
  const int lane = threadIdx.x & 31;
  const int wid  = threadIdx.x >> 5;
  const int lrow = lane & 15;
  const int lhi  = lane >> 4;
  const int gid  = blockIdx.x * 8 + wid;
  const int m0   = (gid >> 4) * 32;
  const int n0   = (gid & 15) * 64;

  const __bf16* arow0 = A + (size_t)(m0 + lrow) * EMBED + lhi * 16;
  const __bf16* arow1 = arow0 + (size_t)16 * EMBED;

  v8f acc[2][4];
  #pragma unroll
  for (int t = 0; t < 4; ++t) {
    const float bv = bias[n0 + t * 16 + lrow];
    #pragma unroll
    for (int r = 0; r < 8; ++r) { acc[0][t][r] = bv; acc[1][t][r] = bv; }
  }

  for (int k0 = 0; k0 < EMBED; k0 += 32) {
    v16bf af0, af1;
    {
      v8bf a0 = *(const v8bf*)(arow0 + k0);
      v8bf a1 = *(const v8bf*)(arow0 + k0 + 8);
      v8bf a2 = *(const v8bf*)(arow1 + k0);
      v8bf a3 = *(const v8bf*)(arow1 + k0 + 8);
      #pragma unroll
      for (int i = 0; i < 8; ++i) {
        af0[i] = a0[i]; af0[8 + i] = a1[i];
        af1[i] = a2[i]; af1[8 + i] = a3[i];
      }
    }
    #pragma unroll
    for (int t = 0; t < 4; ++t) {
      const float* wrow = W + (size_t)(n0 + t * 16 + lrow) * EMBED + k0 + lhi * 16;
      v16bf bf = cvt_frag_f32(*(const v4f*)(wrow),
                              *(const v4f*)(wrow + 4),
                              *(const v4f*)(wrow + 8),
                              *(const v4f*)(wrow + 12));
      acc[0][t] = wmma_bf16(af0, bf, acc[0][t]);
      acc[1][t] = wmma_bf16(af1, bf, acc[1][t]);
    }
  }

  #pragma unroll
  for (int mi = 0; mi < 2; ++mi)
    #pragma unroll
    for (int t = 0; t < 4; ++t) {
      const int n = n0 + t * 16 + lrow;
      #pragma unroll
      for (int r = 0; r < 8; ++r) {
        const int m = m0 + mi * 16 + r + 8 * lhi;
        out[(size_t)m * EMBED + n] = acc[mi][t][r];
      }
    }
}

extern "C" void kernel_launch(void* const* d_in, const int* in_sizes, int n_in,
                              void* d_out, int out_size, void* d_ws, size_t ws_size,
                              hipStream_t stream) {
  const float* q  = (const float*)d_in[0];
  const float* k  = (const float*)d_in[1];
  const float* v  = (const float*)d_in[2];
  const float* Wq = (const float*)d_in[3];
  const float* bq = (const float*)d_in[4];
  const float* Wk = (const float*)d_in[5];
  const float* bk = (const float*)d_in[6];
  const float* Wv = (const float*)d_in[7];
  const float* bv = (const float*)d_in[8];
  const float* Wo = (const float*)d_in[9];
  const float* bo = (const float*)d_in[10];
  float* out = (float*)d_out;

  __bf16* Qb  = (__bf16*)d_ws;         // [B,H,S,Dh]
  __bf16* Kb  = Qb  + HSZ;             // [B,H,S,Dh]
  __bf16* Vtb = Kb  + HSZ;             // [B,H,Dh,S]
  __bf16* Ab  = Vtb + HSZ;             // [B,S,E]

  const dim3 grid(512), block(256);
  const float qscale = 0.125f;         // HEAD_DIM^-0.5, folded into Q

  proj_gemm<0><<<grid, block, 0, stream>>>(q, Wq, bq, Qb, qscale);
  proj_gemm<0><<<grid, block, 0, stream>>>(k, Wk, bk, Kb, 1.0f);
  proj_gemm<1><<<grid, block, 0, stream>>>(v, Wv, bv, Vtb, 1.0f);
  flash_attn<<<grid, block, 0, stream>>>(Qb, Kb, Vtb, Ab);
  out_gemm<<<grid, block, 0, stream>>>(Ab, Wo, bo, out);
}
